// SparseMoE_3874060501223
// MI455X (gfx1250) — compile-verified, matrix operand order checked by reading, not run on hardware
//
#include <hip/hip_runtime.h>
#include <math.h>

#define B_TOK 8192
#define D_DIM 512
#define H_DIM 2048
#define E_NUM 8
#define MAXPE 8192          // each token contributes at most 1 slot per expert
#define MT    32            // token rows per expert tile
#define HCB   128           // H chunk per iteration (8 waves x 16 cols)
#define KC1   64            // k-chunk for GEMM1

typedef __attribute__((ext_vector_type(16))) __bf16 v16bf;
typedef __attribute__((ext_vector_type(8)))  float  v8f;
typedef __attribute__((ext_vector_type(4)))  unsigned int v4u;

union BFrag { v16bf v; v4u q[2]; unsigned short u[16]; };
union F8    { v8f v; float f[8]; };

__device__ __forceinline__ unsigned short f2bf(float f) {
    unsigned int b = __float_as_uint(f);
    unsigned int r = b + 0x7FFFu + ((b >> 16) & 1u);   // round-to-nearest-even
    return (unsigned short)(r >> 16);
}

// ---------------------------------------------------------------- init
__global__ void moe_init(int* cnt) {
    if (threadIdx.x < E_NUM) cnt[threadIdx.x] = 0;
}

// ---------------------------------------------------------------- weight cvt
__global__ __launch_bounds__(256)
void moe_cvtw(const float* __restrict__ src, unsigned int* __restrict__ dst, int n2) {
    int i = blockIdx.x * 256 + threadIdx.x;
    if (i < n2) {
        const float2 ab = ((const float2*)src)[i];
        dst[i] = (unsigned int)f2bf(ab.x) | ((unsigned int)f2bf(ab.y) << 16);
    }
}

// ---------------------------------------------------------------- gating
// x tile staged via CDNA5 async global->LDS DMA (ASYNCcnt), no VGPR round-trip.
__global__ __launch_bounds__(256)
void moe_gate(const float* __restrict__ x, const float* __restrict__ Wg,
              const float* __restrict__ bg,
              float* __restrict__ wts, int* __restrict__ list,
              int* __restrict__ cnt, int* __restrict__ topi) {
    __shared__ __align__(16) float xsh[32][D_DIM];   // unpadded: layout-preserving copy
    __shared__ float wgs[D_DIM * E_NUM];
    __shared__ float lg[32][E_NUM];

    const int tb = blockIdx.x * 32;
    // async copy: 32*512 floats = 4096 x b128, 16 per thread
    for (int i = threadIdx.x; i < (32 * D_DIM) / 4; i += 256) {
        int e4 = i * 4;                               // element index, 16B aligned
        unsigned lds = (unsigned)(uintptr_t)&xsh[0][0] + (unsigned)(e4 * 4);
        const float* g = &x[(size_t)tb * D_DIM + e4];
        asm volatile("global_load_async_to_lds_b128 %0, %1, off"
                     :: "v"(lds), "v"(g) : "memory");
    }
    for (int i = threadIdx.x; i < D_DIM * E_NUM; i += 256) wgs[i] = Wg[i];
    asm volatile("s_wait_asynccnt 0" ::: "memory");
    __syncthreads();

    const int tl = threadIdx.x >> 3;
    const int e  = threadIdx.x & 7;
    float acc = bg[e];
    #pragma unroll 8
    for (int d = 0; d < D_DIM; d++) acc += xsh[tl][d] * wgs[d * E_NUM + e];
    lg[tl][e] = acc;
    __syncthreads();

    if (threadIdx.x < 32) {
        const int tok = tb + threadIdx.x;
        float v0 = -3.4e38f; int i0 = 0;
        for (int ee = 0; ee < E_NUM; ee++) {
            float v = lg[threadIdx.x][ee];
            if (v > v0) { v0 = v; i0 = ee; }
        }
        float v1 = -3.4e38f; int i1 = 0;
        for (int ee = 0; ee < E_NUM; ee++) {
            if (ee == i0) continue;
            float v = lg[threadIdx.x][ee];
            if (v > v1) { v1 = v; i1 = ee; }
        }
        float p1 = __expf(v1 - v0);        // v0 >= v1
        float s  = 1.0f + p1;
        topi[tok * 2 + 0] = i0;
        topi[tok * 2 + 1] = i1;
        wts[tok * 2 + 0] = 1.0f / s;
        wts[tok * 2 + 1] = p1 / s;
        int p0 = atomicAdd(&cnt[i0], 1);
        list[i0 * MAXPE + p0] = tok * 2 + 0;
        int p1i = atomicAdd(&cnt[i1], 1);
        list[i1 * MAXPE + p1i] = tok * 2 + 1;
    }
}

// ---------------------------------------------------------------- expert FFN
// Block = 256 threads = 8 wave32s; 32 gathered tokens for one expert.
// Weight slabs staged TRANSPOSED ([N][K]) so B-fragments load as 2x b128.
__global__ __launch_bounds__(256)
void moe_expert(const float* __restrict__ x,
                const unsigned short* __restrict__ W1bf,
                const float* __restrict__ b1,
                const unsigned short* __restrict__ W2bf,
                const float* __restrict__ b2,
                const int* __restrict__ cnt, const int* __restrict__ list,
                const float* __restrict__ wts, float* __restrict__ ybuf) {
    const int e = blockIdx.x >> 8;               // 256 tiles per expert
    const int t = blockIdx.x & 255;
    const int n = cnt[e];
    if (t * MT >= n) return;
    const int rem = min(MT, n - t * MT);

    __shared__ __align__(16) unsigned short xs[MT][D_DIM + 8];   // A1  (row 1040B)
    __shared__ __align__(16) unsigned short w1t[HCB][KC1 + 8];   // B1^T (row 144B)
    __shared__ __align__(16) unsigned short hs[MT][HCB + 8];     // A2  (row 272B)
    __shared__ __align__(16) unsigned short w2t[D_DIM][40];      // B2^T (row 80B)
    __shared__ int   rslot[MT];
    __shared__ float rwgt[MT];

    const int tid = threadIdx.x;
    if (tid < MT) {
        int gi = t * MT + tid;
        int se = (gi < n) ? list[e * MAXPE + gi] : 0;
        rslot[tid] = se;
        rwgt[tid]  = (gi < n) ? wts[se] : 0.0f;
    }
    __syncthreads();

    // stage x rows -> bf16, dword-packed stores
    for (int i = tid; i < MT * D_DIM / 2; i += 256) {
        int r = i >> 8, cp = i & 255;
        int tok = rslot[r] >> 1;
        const float* p = &x[(size_t)tok * D_DIM + 2 * cp];
        unsigned int lo = f2bf(p[0]), hi = f2bf(p[1]);
        *(unsigned int*)&xs[r][2 * cp] = lo | (hi << 16);
    }
    __syncthreads();

    const int w  = tid >> 5;     // wave 0..7 (wave32)
    const int l  = tid & 31;
    const int g  = l >> 4;       // half-wave
    const int ln = l & 15;

    F8 yacc[2][4];
    #pragma unroll
    for (int mt = 0; mt < 2; mt++)
        #pragma unroll
        for (int j = 0; j < 4; j++)
            #pragma unroll
            for (int r = 0; r < 8; r++) yacc[mt][j].f[r] = 0.0f;

    for (int hc = 0; hc < H_DIM; hc += HCB) {
        // ---- GEMM1: h[32,128] = x[32,512] @ W1_e[512, hc:hc+128]
        F8 hacc[2];
        #pragma unroll
        for (int mt = 0; mt < 2; mt++)
            #pragma unroll
            for (int r = 0; r < 8; r++) hacc[mt].f[r] = 0.0f;

        for (int k = 0; k < D_DIM; k += KC1) {
            // stage W1 slab transposed: w1t[c][kk] = W1[k+kk][hc+c]
            for (int i = tid; i < (HCB / 2) * KC1; i += 256) {
                int cp = i & (HCB / 2 - 1);       // col pair 0..63
                int kk = i >> 6;                  // 0..63
                unsigned int v = *(const unsigned int*)
                    &W1bf[((size_t)e * D_DIM + k + kk) * H_DIM + hc + 2 * cp];
                w1t[2 * cp][kk]     = (unsigned short)v;
                w1t[2 * cp + 1][kk] = (unsigned short)(v >> 16);
            }
            __syncthreads();
            // prefetch next W1 slab rows (L2-resident) while computing this one
            if (k + KC1 < D_DIM && tid < 2 * KC1) {
                int rr = tid >> 1, half = tid & 1;
                __builtin_prefetch(
                    &W1bf[((size_t)e * D_DIM + k + KC1 + rr) * H_DIM + hc + half * 64],
                    0, 1);
            }
            #pragma unroll
            for (int ks = 0; ks < KC1; ks += 32) {
                BFrag bf;
                bf.q[0] = *(const v4u*)&w1t[w * 16 + ln][ks + g * 8];
                bf.q[1] = *(const v4u*)&w1t[w * 16 + ln][ks + 16 + g * 8];
                #pragma unroll
                for (int mt = 0; mt < 2; mt++) {
                    BFrag af;
                    af.q[0] = *(const v4u*)&xs[mt * 16 + ln][k + ks + g * 8];
                    af.q[1] = *(const v4u*)&xs[mt * 16 + ln][k + ks + 16 + g * 8];
                    hacc[mt].v = __builtin_amdgcn_wmma_f32_16x16x32_bf16(
                        false, af.v, false, bf.v, (short)0, hacc[mt].v, false, false);
                }
            }
            __syncthreads();
        }
        // bias + exact GELU -> bf16 LDS
        const float bb = b1[(size_t)e * H_DIM + hc + w * 16 + ln];
        #pragma unroll
        for (int mt = 0; mt < 2; mt++)
            #pragma unroll
            for (int r = 0; r < 8; r++) {
                float v = hacc[mt].f[r] + bb;
                v = 0.5f * v * (1.0f + erff(v * 0.70710678118f));
                hs[mt * 16 + g * 8 + r][w * 16 + ln] = f2bf(v);
            }
        __syncthreads();

        // ---- GEMM2: y[32,512] += h[32,128] @ W2_e[hc:hc+128, 0:512]
        for (int k2 = 0; k2 < HCB; k2 += 32) {
            // stage W2 slab transposed: w2t[c][kk] = W2[hc+k2+kk][c]
            for (int i = tid; i < (D_DIM / 2) * 32; i += 256) {
                int cp = i & (D_DIM / 2 - 1);     // col pair 0..255
                int kk = i >> 8;                  // 0..31
                unsigned int v = *(const unsigned int*)
                    &W2bf[((size_t)e * H_DIM + hc + k2 + kk) * D_DIM + 2 * cp];
                w2t[2 * cp][kk]     = (unsigned short)v;
                w2t[2 * cp + 1][kk] = (unsigned short)(v >> 16);
            }
            __syncthreads();
            // prefetch next W2 slab (32 rows x 1KB) while computing this one
            if (k2 + 32 < HCB) {
                int rr = tid >> 3, seg = tid & 7;
                __builtin_prefetch(
                    &W2bf[((size_t)e * H_DIM + hc + k2 + 32 + rr) * D_DIM + seg * 64],
                    0, 1);
            }
            #pragma unroll
            for (int mt = 0; mt < 2; mt++) {
                BFrag af;
                af.q[0] = *(const v4u*)&hs[mt * 16 + ln][k2 + g * 8];
                af.q[1] = *(const v4u*)&hs[mt * 16 + ln][k2 + 16 + g * 8];
                #pragma unroll
                for (int j = 0; j < 4; j++) {
                    BFrag bf;
                    bf.q[0] = *(const v4u*)&w2t[w * 64 + j * 16 + ln][g * 8];
                    bf.q[1] = *(const v4u*)&w2t[w * 64 + j * 16 + ln][16 + g * 8];
                    yacc[mt][j].v = __builtin_amdgcn_wmma_f32_16x16x32_bf16(
                        false, af.v, false, bf.v, (short)0, yacc[mt][j].v, false, false);
                }
            }
            __syncthreads();
        }
    }

    // epilogue: +b2, scale by gate weight, scatter to ybuf[slot]
    #pragma unroll
    for (int j = 0; j < 4; j++) {
        const int col = w * 64 + j * 16 + ln;
        const float bz = b2[(size_t)e * D_DIM + col];
        #pragma unroll
        for (int mt = 0; mt < 2; mt++)
            #pragma unroll
            for (int r = 0; r < 8; r++) {
                int row = mt * 16 + g * 8 + r;
                if (row < rem)
                    ybuf[(size_t)rslot[row] * D_DIM + col] =
                        (yacc[mt][j].f[r] + bz) * rwgt[row];
            }
    }
}

// ---------------------------------------------------------------- combine
__global__ __launch_bounds__(256)
void moe_combine(const float* __restrict__ ybuf, float* __restrict__ out) {
    size_t i = (size_t)blockIdx.x * 256 + threadIdx.x;   // over B*D
    int tok = (int)(i >> 9);
    int col = (int)(i & (D_DIM - 1));
    out[i] = ybuf[(size_t)(tok * 2) * D_DIM + col] +
             ybuf[(size_t)(tok * 2 + 1) * D_DIM + col];
}

// ---------------------------------------------------------------- launch
extern "C" void kernel_launch(void* const* d_in, const int* in_sizes, int n_in,
                              void* d_out, int out_size, void* d_ws, size_t ws_size,
                              hipStream_t stream) {
    const float* x  = (const float*)d_in[0];
    const float* Wg = (const float*)d_in[1];
    const float* bg = (const float*)d_in[2];
    const float* W1 = (const float*)d_in[3];
    const float* b1 = (const float*)d_in[4];
    const float* W2 = (const float*)d_in[5];
    const float* b2 = (const float*)d_in[6];

    float* out  = (float*)d_out;
    int*   topi = (int*)((float*)d_out + (size_t)B_TOK * D_DIM);

    // workspace layout
    const size_t W_ELT = (size_t)E_NUM * D_DIM * H_DIM;     // 8.39M per matrix
    char* ws = (char*)d_ws;
    int*   cnt  = (int*)ws;                                             // 8 ints
    float* wts  = (float*)(ws + 256);                                   // B*2 f32
    int*   list = (int*)(ws + 256 + (size_t)B_TOK * 2 * 4);             // E*MAXPE
    float* ybuf = (float*)(ws + 256 + (size_t)B_TOK * 2 * 4
                              + (size_t)E_NUM * MAXPE * 4);             // B*2*D f32
    unsigned short* W1bf = (unsigned short*)((char*)ybuf
                              + (size_t)B_TOK * 2 * D_DIM * 4);         // bf16
    unsigned short* W2bf = W1bf + W_ELT;

    moe_init<<<1, 64, 0, stream>>>(cnt);
    moe_cvtw<<<(int)(W_ELT / 2 / 256), 256, 0, stream>>>(
        W1, (unsigned int*)W1bf, (int)(W_ELT / 2));
    moe_cvtw<<<(int)(W_ELT / 2 / 256), 256, 0, stream>>>(
        W2, (unsigned int*)W2bf, (int)(W_ELT / 2));
    moe_gate<<<B_TOK / 32, 256, 0, stream>>>(x, Wg, bg, wts, list, cnt, topi);
    moe_expert<<<E_NUM * (MAXPE / MT), 256, 0, stream>>>(
        x, W1bf, b1, W2bf, b2, cnt, list, wts, ybuf);
    moe_combine<<<(B_TOK * D_DIM) / 256, 256, 0, stream>>>(ybuf, out);
}
